// SelfAttention_69183333203999
// MI455X (gfx1250) — matrix operand order repaired and checked by reading, not compiled, not run on hardware
//
#include <hip/hip_runtime.h>
#include <hip/hip_bf16.h>
#include <stdint.h>

// Problem constants (match reference)
#define B_ 8
#define N_ 2048
#define D_ 1024
#define NEG_ (-1e9f)

typedef __bf16 bf16_t;
typedef __attribute__((ext_vector_type(4)))  bf16_t v4bf;
typedef __attribute__((ext_vector_type(8)))  bf16_t v8bf;
typedef __attribute__((ext_vector_type(16))) bf16_t v16bf;
typedef __attribute__((ext_vector_type(4)))  float  v4f;
typedef __attribute__((ext_vector_type(8)))  float  v8f;

__device__ __forceinline__ bf16_t f2bf(float f) {
  // round-to-nearest-even fp32 -> bf16
  uint32_t u = __builtin_bit_cast(uint32_t, f);
  uint32_t r = u + 0x7FFFu + ((u >> 16) & 1u);
  uint16_t s = (uint16_t)(r >> 16);
  return __builtin_bit_cast(bf16_t, s);
}

// ---- gfx1250 async-to-LDS helpers -----------------------------------------
// Raw LDS byte offset of a generic pointer that points into __shared__.
typedef __attribute__((address_space(3))) char lds_char;
__device__ __forceinline__ uint32_t lds_off(const void* p) {
  return (uint32_t)(uintptr_t)(lds_char*)(void*)p;
}

// GLOBAL_LOAD_ASYNC_TO_LDS_B128: VDST = LDS byte address (VGPR),
// VADDR = 64-bit global address. Tracked with ASYNCcnt.
__device__ __forceinline__ void async_copy_b128(uint32_t ldsaddr, const void* gaddr) {
  asm volatile("global_load_async_to_lds_b128 %0, %1, off"
               :: "v"(ldsaddr), "v"(gaddr)
               : "memory");
}

__device__ __forceinline__ void wait_async0() {
  asm volatile("s_wait_asynccnt 0x0" ::: "memory");
}

// ---------------------------------------------------------------------------
// fp32 -> bf16 convert (optionally scaled; folds the 0.06 Q-scale into Wq)
// ---------------------------------------------------------------------------
__global__ __launch_bounds__(256)
void cvt_f32_bf16(const float* __restrict__ in, bf16_t* __restrict__ out,
                  size_t n4, float scale) {
  size_t i = ((size_t)blockIdx.x * 256 + threadIdx.x);
  if (i >= n4) return;
  size_t e = i * 4;
  v4f v = *(const v4f*)(in + e);
  v4bf o;
  o[0] = f2bf(v[0] * scale);
  o[1] = f2bf(v[1] * scale);
  o[2] = f2bf(v[2] * scale);
  o[3] = f2bf(v[3] * scale);
  *(v4bf*)(out + e) = o;
}

// ---------------------------------------------------------------------------
// NT GEMM:  C[M x Nn] = A[M x K] * Bsrc[Nn x K]^T   (A, Bsrc row-major bf16)
// 128 threads = 4 waves, block tile 64x64, wave tile 32x32 (2x2 wmma frags).
// K-chunk 64 double-buffered through LDS with async-to-LDS copies.
// ---------------------------------------------------------------------------
enum { EPI_BF16 = 0, EPI_BF16_T = 1, EPI_MASK_F32 = 2, EPI_F32 = 3 };

#define KC_    64
#define LDSTR_ 72   // padded row stride (elements): 144 B, 16B aligned, bank-spread

template <int EPI>
__global__ __launch_bounds__(128)
void gemm_nt(const bf16_t* __restrict__ Abase, const bf16_t* __restrict__ Bbase,
             size_t aStride, size_t bStride,
             void* __restrict__ outBase, size_t oStride,
             int lda, int ldb, int ldo, int K,
             const unsigned char* __restrict__ maskBase) {
  __shared__ __align__(16) bf16_t As[2][64 * LDSTR_];
  __shared__ __align__(16) bf16_t Bs[2][64 * LDSTR_];

  const int z    = blockIdx.z;
  const bf16_t* A  = Abase + (size_t)z * aStride;
  const bf16_t* Bm = Bbase + (size_t)z * bStride;

  const int t    = threadIdx.x;
  const int lane = t & 31;
  const int wave = t >> 5;
  const int m    = lane & 15;
  const int h    = lane >> 4;
  const int wm32 = (wave >> 1) * 32;       // wave row offset inside block tile
  const int wn32 = (wave & 1) * 32;        // wave col offset inside block tile
  const int rowBase = blockIdx.x * 64;
  const int colBase = blockIdx.y * 64;

  // Stage one 64 x KC_ tile of A and of B into LDS buffer `buf` (async).
  // 64 rows * 64 elems * 2B = 8 KB per matrix = 512 b128 chunks -> 4/thread.
  auto stage = [&](int buf, int k) {
#pragma unroll
    for (int it = 0; it < 4; ++it) {
      const int q = t + 128 * it;          // 0..511
      const int r = q >> 3;                // row 0..63
      const int c = (q & 7) * 8;           // element col {0,8,...,56}
      async_copy_b128(lds_off(&As[buf][r * LDSTR_ + c]),
                      A + (size_t)(rowBase + r) * lda + k + c);
      async_copy_b128(lds_off(&Bs[buf][r * LDSTR_ + c]),
                      Bm + (size_t)(colBase + r) * ldb + k + c);
    }
  };

  v8f acc00 = {}, acc01 = {}, acc10 = {}, acc11 = {};

  stage(0, 0);
  wait_async0();
  __syncthreads();

  const int nk = K / KC_;
  for (int kc = 0; kc < nk; ++kc) {
    const int cur = kc & 1;
    if (kc + 1 < nk) stage(cur ^ 1, (kc + 1) * KC_);   // overlap with compute

    const bf16_t* as = &As[cur][0];
    const bf16_t* bs = &Bs[cur][0];
#pragma unroll
    for (int ks = 0; ks < KC_; ks += 32) {
      // A frag (16-bit A 16x32 layout): lane holds K chunks [8h,+8) U [16+8h,+8)
      const bf16_t* ar0 = as + (wm32 + m)      * LDSTR_ + ks + 8 * h;
      const bf16_t* ar1 = as + (wm32 + 16 + m) * LDSTR_ + ks + 8 * h;
      v8bf a0lo = *(const v8bf*)(ar0);
      v8bf a0hi = *(const v8bf*)(ar0 + 16);
      v8bf a1lo = *(const v8bf*)(ar1);
      v8bf a1hi = *(const v8bf*)(ar1 + 16);
      v16bf a0 = __builtin_shufflevector(a0lo, a0hi, 0,1,2,3,4,5,6,7,8,9,10,11,12,13,14,15);
      v16bf a1 = __builtin_shufflevector(a1lo, a1hi, 0,1,2,3,4,5,6,7,8,9,10,11,12,13,14,15);
      // B frag (32x16): lane = column, K chunks [8h,+8) U [16+8h,+8)
      const bf16_t* br0 = bs + (wn32 + m)      * LDSTR_ + ks + 8 * h;
      const bf16_t* br1 = bs + (wn32 + 16 + m) * LDSTR_ + ks + 8 * h;
      v16bf b0 = __builtin_shufflevector(*(const v8bf*)(br0), *(const v8bf*)(br0 + 16), 0,1,2,3,4,5,6,7,8,9,10,11,12,13,14,15);
      v16bf b1 = __builtin_shufflevector(*(const v8bf*)(br1), *(const v8bf*)(br1 + 16), 0,1,2,3,4,5,6,7,8,9,10,11,12,13,14,15);

      acc00 = __builtin_amdgcn_wmma_f32_16x16x32_bf16(false, a0, false, b0, (short)0, acc00, false, false);
      acc01 = __builtin_amdgcn_wmma_f32_16x16x32_bf16(false, a0, false, b1, (short)0, acc01, false, false);
      acc10 = __builtin_amdgcn_wmma_f32_16x16x32_bf16(false, a1, false, b0, (short)0, acc10, false, false);
      acc11 = __builtin_amdgcn_wmma_f32_16x16x32_bf16(false, a1, false, b1, (short)0, acc11, false, false);
    }

    wait_async0();      // next chunk fully in LDS (our own async ops)
    __syncthreads();    // all waves' async copies visible; safe to swap buffers
  }

  // Epilogue. C layout: VGPR r, lane l -> row = r + 8*(l>>4), col = l&15
  const unsigned char* maskRow = (EPI == EPI_MASK_F32) ? (maskBase + (size_t)z * N_) : nullptr;

  const v8f accs[4] = {acc00, acc01, acc10, acc11};
#pragma unroll
  for (int ti = 0; ti < 2; ++ti) {
#pragma unroll
    for (int tj = 0; tj < 2; ++tj) {
      const v8f& acc = accs[ti * 2 + tj];
      const int col = colBase + wn32 + 16 * tj + m;
#pragma unroll
      for (int r = 0; r < 8; ++r) {
        const int row = rowBase + wm32 + 16 * ti + 8 * h + r;
        float v = acc[r];
        if constexpr (EPI == EPI_BF16) {
          bf16_t* o = (bf16_t*)outBase + (size_t)z * oStride;
          o[(size_t)row * ldo + col] = f2bf(v);
        } else if constexpr (EPI == EPI_BF16_T) {
          bf16_t* o = (bf16_t*)outBase + (size_t)z * oStride;
          o[(size_t)col * ldo + row] = f2bf(v);   // transposed store (Vt)
        } else if constexpr (EPI == EPI_MASK_F32) {
          float* o = (float*)outBase + (size_t)z * oStride;
          bool mm = maskRow[row] && maskRow[col];
          o[(size_t)row * ldo + col] = mm ? v : NEG_;
        } else {
          float* o = (float*)outBase + (size_t)z * oStride;
          o[(size_t)row * ldo + col] = v;
        }
      }
    }
  }
}

// ---------------------------------------------------------------------------
// Row softmax over N_=2048 fp32 (in place in d_out att region) and emit
// P = att^T in bf16 (reference computes y = einsum('bji,bjd->bid', att, V))
// ---------------------------------------------------------------------------
__global__ __launch_bounds__(256)
void softmax_rows(float* __restrict__ att, bf16_t* __restrict__ P) {
  const int i = blockIdx.x;   // row within batch
  const int b = blockIdx.y;
  float* row = att + ((size_t)b * N_ + i) * (size_t)N_;
  bf16_t* Pb = P + (size_t)b * N_ * (size_t)N_;

  const int t = threadIdx.x;
  const int lane = t & 31, w = t >> 5;
  __shared__ float red[8];
  __shared__ float bcast;

  float v[8];
  v4f x0 = *(const v4f*)(row + t * 8);
  v4f x1 = *(const v4f*)(row + t * 8 + 4);
#pragma unroll
  for (int e = 0; e < 4; ++e) { v[e] = x0[e]; v[4 + e] = x1[e]; }

  // --- max reduce ---
  float mx = -3.4e38f;
#pragma unroll
  for (int e = 0; e < 8; ++e) mx = fmaxf(mx, v[e]);
  for (int o = 16; o > 0; o >>= 1) mx = fmaxf(mx, __shfl_xor(mx, o, 32));
  if (lane == 0) red[w] = mx;
  __syncthreads();
  if (t == 0) {
    float x = red[0];
#pragma unroll
    for (int q = 1; q < 8; ++q) x = fmaxf(x, red[q]);
    bcast = x;
  }
  __syncthreads();
  mx = bcast;
  __syncthreads();

  // --- exp + sum reduce ---
  float s = 0.f;
#pragma unroll
  for (int e = 0; e < 8; ++e) { v[e] = __expf(v[e] - mx); s += v[e]; }
  for (int o = 16; o > 0; o >>= 1) s += __shfl_xor(s, o, 32);
  if (lane == 0) red[w] = s;
  __syncthreads();
  if (t == 0) {
    float x = 0.f;
#pragma unroll
    for (int q = 0; q < 8; ++q) x += red[q];
    bcast = x;
  }
  __syncthreads();
  const float inv = 1.0f / bcast;

  v4f o0, o1;
#pragma unroll
  for (int e = 0; e < 4; ++e) { o0[e] = v[e] * inv; o1[e] = v[4 + e] * inv; }
  *(v4f*)(row + t * 8)     = o0;
  *(v4f*)(row + t * 8 + 4) = o1;
#pragma unroll
  for (int e = 0; e < 8; ++e) {
    int j = t * 8 + e;
    Pb[(size_t)j * N_ + i] = f2bf(v[e] * inv);   // P[j][i] = att[i][j]
  }
}

// ---------------------------------------------------------------------------
// Host-side orchestration
// ---------------------------------------------------------------------------
extern "C" void kernel_launch(void* const* d_in, const int* in_sizes, int n_in,
                              void* d_out, int out_size, void* d_ws, size_t ws_size,
                              hipStream_t stream) {
  (void)in_sizes; (void)n_in; (void)out_size; (void)ws_size;

  const float*         x    = (const float*)d_in[0];
  const unsigned char* mask = (const unsigned char*)d_in[1];  // bool array, 1 byte/elem
  const float*         Wq   = (const float*)d_in[2];
  const float*         Wk   = (const float*)d_in[3];
  const float*         Wv   = (const float*)d_in[4];
  const float*         Wo   = (const float*)d_in[5];

  float* y_out   = (float*)d_out;                       // [B,N,D]
  float* att_out = y_out + (size_t)B_ * N_ * D_;        // [B,N,N]

  // workspace carve-up (bf16 buffers)
  char* ws = (char*)d_ws;
  size_t off = 0;
  auto carve = [&](size_t bytes) -> void* {
    void* p = ws + off;
    off += (bytes + 255) & ~(size_t)255;
    return p;
  };
  const size_t BND = (size_t)B_ * N_ * D_;
  const size_t BNN = (size_t)B_ * N_ * (size_t)N_;
  bf16_t* xbf  = (bf16_t*)carve(BND * 2);
  bf16_t* wqb  = (bf16_t*)carve((size_t)D_ * D_ * 2);
  bf16_t* wkb  = (bf16_t*)carve((size_t)D_ * D_ * 2);
  bf16_t* wvb  = (bf16_t*)carve((size_t)D_ * D_ * 2);
  bf16_t* wob  = (bf16_t*)carve((size_t)D_ * D_ * 2);
  bf16_t* qbf  = (bf16_t*)carve(BND * 2);
  bf16_t* kbf  = (bf16_t*)carve(BND * 2);
  bf16_t* vt   = (bf16_t*)carve(BND * 2);   // V transposed: [B, D, N]
  bf16_t* pbuf = (bf16_t*)carve(BNN * 2);   // att^T bf16:   [B, N, N]
  bf16_t* ymid = (bf16_t*)carve(BND * 2);   // att^T @ V:    [B, N, D]

  // 1) converts
  {
    size_t n4 = BND / 4;
    cvt_f32_bf16<<<(unsigned)((n4 + 255) / 256), 256, 0, stream>>>(x, xbf, n4, 1.0f);
    size_t w4 = (size_t)D_ * D_ / 4;
    unsigned gw = (unsigned)((w4 + 255) / 256);
    cvt_f32_bf16<<<gw, 256, 0, stream>>>(Wq, wqb, w4, 0.06f);  // fold Q scale
    cvt_f32_bf16<<<gw, 256, 0, stream>>>(Wk, wkb, w4, 1.0f);
    cvt_f32_bf16<<<gw, 256, 0, stream>>>(Wv, wvb, w4, 1.0f);
    cvt_f32_bf16<<<gw, 256, 0, stream>>>(Wo, wob, w4, 1.0f);
  }

  dim3 blk(128);
  const size_t ND = (size_t)N_ * D_;
  const size_t NN = (size_t)N_ * (size_t)N_;
  const size_t DN = (size_t)D_ * (size_t)N_;

  // 2) Q = x Wq^T, K = x Wk^T (bf16), Vt = (x Wv^T)^T (bf16)
  dim3 gProj(N_ / 64, D_ / 64, B_);
  gemm_nt<EPI_BF16>  <<<gProj, blk, 0, stream>>>(xbf, wqb, ND, 0, qbf, ND, D_, D_, D_, D_, nullptr);
  gemm_nt<EPI_BF16>  <<<gProj, blk, 0, stream>>>(xbf, wkb, ND, 0, kbf, ND, D_, D_, D_, D_, nullptr);
  gemm_nt<EPI_BF16_T><<<gProj, blk, 0, stream>>>(xbf, wvb, ND, 0, vt,  DN, D_, D_, N_, D_, nullptr);

  // 3) masked logits = Q K^T  (fp32, straight into d_out att region)
  dim3 gLog(N_ / 64, N_ / 64, B_);
  gemm_nt<EPI_MASK_F32><<<gLog, blk, 0, stream>>>(qbf, kbf, ND, ND, att_out, NN, D_, D_, N_, D_, mask);

  // 4) softmax rows in place; emit P = att^T bf16
  softmax_rows<<<dim3(N_, B_), 256, 0, stream>>>(att_out, pbuf);

  // 5) y_mid = P Vt^T = att^T V   (bf16)
  dim3 gAV(N_ / 64, D_ / 64, B_);
  gemm_nt<EPI_BF16><<<gAV, blk, 0, stream>>>(pbuf, vt, NN, DN, ymid, ND, N_, N_, D_, N_, nullptr);

  // 6) y = y_mid Wo^T  (fp32 -> d_out)
  gemm_nt<EPI_F32><<<gAV, blk, 0, stream>>>(ymid, wob, ND, 0, y_out, ND, D_, D_, D_, D_, nullptr);
}
